// GPT_88124138979768
// MI455X (gfx1250) — matrix operand order repaired and checked by reading, not compiled, not run on hardware
//
#include <hip/hip_runtime.h>

// Model dims (match reference)
constexpr int L_  = 6;
constexpr int H_  = 16;
constexpr int DM_ = 1024;
constexpr int DH_ = 64;
constexpr int DFF_ = 4096;
constexpr int V_  = 32000;
constexpr int B_  = 2;
constexpr int S_  = 1024;

typedef __attribute__((ext_vector_type(16))) __bf16 v16bf;
typedef __attribute__((ext_vector_type(8)))  __bf16 v8bf;
typedef __attribute__((ext_vector_type(8)))  float  v8f;

__device__ __forceinline__ __bf16 f2bf(float f) { return (__bf16)f; }

// ---------------------------------------------------------------------------
// WMMA bf16 fragment layouts (CDNA5 ISA 7.12.2):
//   A (16x32, MxK): lane = m + 16*((k>>3)&1), slot = (k>>4)*8 + (k&7)
//   B (32x16, KxN): lane = n + 16*((k>>3)&1), slot = (k>>4)*8 + (k&7)
// BOTH operands are kept pre-packed in fragment-major bf16 in global memory:
//   - weights: repacked once per GEMM (repack_w_kernel)
//   - activations: written packed by their producer (embed / LN / epilogues)
// so the GEMM inner loop is pure {global_load_b128 x10, wmma x4}: no LDS,
// no barriers, no conversions.
// ---------------------------------------------------------------------------

// A-fragment-major offset for element (m,k) of an [M,Kc] activation:
__device__ __forceinline__ long apack_off(int m, int k, int Kc) {
  return (((long)(m >> 4) * (Kc >> 5) + (k >> 5)) * 32 + (m & 15) +
          (((k >> 3) & 1) << 4)) * 16 + (((k >> 4) & 1) << 3) + (k & 7);
}

// -------------------------------------------------------------------------
// Repack f32 weight [K,N] (row-major, batched) -> B-fragment-major bf16.
// grid: (ceil(N/256), K/4, nbatch). Thread: one n, 4 consecutive k.
// -------------------------------------------------------------------------
__global__ void repack_w_kernel(const float* __restrict__ W,
                                __bf16* __restrict__ Wp,
                                int N, long sWb, long sWp) {
  const int n = blockIdx.x * 256 + threadIdx.x;
  if (n >= N) return;
  const int k0 = blockIdx.y << 2;
  const float* Ws = W + (long)blockIdx.z * sWb;
  __bf16* Wd = Wp + (long)blockIdx.z * sWp;
  float f0 = Ws[(long)(k0 + 0) * N + n];
  float f1 = Ws[(long)(k0 + 1) * N + n];
  float f2 = Ws[(long)(k0 + 2) * N + n];
  float f3 = Ws[(long)(k0 + 3) * N + n];
  union { __bf16 h[4]; uint2 u2; } p;
  p.h[0] = f2bf(f0); p.h[1] = f2bf(f1); p.h[2] = f2bf(f2); p.h[3] = f2bf(f3);
  const int kt = k0 >> 5;
  const int lane = (n & 15) + (((k0 >> 3) & 1) << 4);
  const long off = (((long)kt * (N >> 4) + (n >> 4)) * 32 + lane) * 16 + (((k0 >> 4) & 1) << 3) + (k0 & 7);
  *(uint2*)&Wd[off] = p.u2;
}

// -------------------------------------------------------------------------
// Embedding + positional encoding: x(f32) and xbf (A-fragment-major bf16).
// 256 thr/row; each thread: 4 consecutive dims -> float4 store + b64 pack.
// -------------------------------------------------------------------------
__global__ void embed_kernel(const int* __restrict__ tokens,
                             const float* __restrict__ emb,
                             float* __restrict__ x,
                             __bf16* __restrict__ xbf) {
  const int row = blockIdx.x;          // 0 .. B*S-1
  const int s = row % S_;
  const int tok = tokens[row];
  const float scl = 32.0f;             // sqrt(1024)
  const float negln = -logf(10000.0f) / (float)DM_;
  const int d0 = threadIdx.x * 4;
  float4 ev = *(const float4*)&emb[(long)tok * DM_ + d0];
  const float fr0 = __expf((float)d0 * negln);          // 2*(d0/2) = d0
  const float fr1 = __expf((float)(d0 + 2) * negln);
  float4 r;
  r.x = ev.x * scl + __sinf((float)s * fr0);
  r.y = ev.y * scl + __cosf((float)s * fr0);
  r.z = ev.z * scl + __sinf((float)s * fr1);
  r.w = ev.w * scl + __cosf((float)s * fr1);
  *(float4*)&x[(long)row * DM_ + d0] = r;
  union { __bf16 h[4]; uint2 u2; } p;
  p.h[0] = f2bf(r.x); p.h[1] = f2bf(r.y); p.h[2] = f2bf(r.z); p.h[3] = f2bf(r.w);
  *(uint2*)&xbf[apack_off(row, d0, DM_)] = p.u2;
}

// -------------------------------------------------------------------------
// Batched GEMM: C = act(A @ W + bias); A and W pre-packed fragment-major.
// Block = 128 thr (4 waves). WMW = waves along M (2: 32x128; 4: 64x64).
// Inner loop: 2 (A) + 8 (W) global_load_b128 + 4 WMMA; no LDS, no barriers.
// FLAGS: bit0 = bias, bit1 = ReLU.
// OMODE: 0 = f32 row-major, 1 = bf16 row-major, 2 = bf16 V-fragment-major,
//        3 = bf16 A-fragment-major (packed for the next GEMM).
// -------------------------------------------------------------------------
template <int WMW, int FLAGS, int OMODE>
__global__ void gemm_bf16_wmma(const __bf16* __restrict__ Apbase,
                               const __bf16* __restrict__ Wpbase,
                               const float* __restrict__ biasBase,
                               void* __restrict__ Cbase,
                               int N, int K, int ldc,
                               long sAb, long sAh, long sWh,
                               long sCb, long sCh, int Hb) {
  constexpr int WNW = 4 / WMW;        // waves along N
  const int z = blockIdx.z;
  const int bb = z / Hb, hh = z % Hb;
  const __bf16* Ap = Apbase + bb * sAb + hh * sAh;
  const __bf16* Wp = Wpbase + hh * sWh;

  const int tid = threadIdx.x;
  const int wave = tid >> 5;
  const int lane = tid & 31;
  const int hf = lane >> 4;
  const int lm = lane & 15;
  const int wm = wave / WNW;
  const int wn = wave % WNW;

  const int mb0 = blockIdx.y * (16 * WMW);
  const int nb0 = blockIdx.x * (64 * WNW);
  const int ntiles16 = N >> 4;
  const int ktiles = K >> 5;

  v8f acc[4];
#pragma unroll
  for (int i = 0; i < 4; ++i) acc[i] = (v8f){0.f,0.f,0.f,0.f,0.f,0.f,0.f,0.f};

  const __bf16* aptr = Ap + ((long)((mb0 >> 4) + wm) * ktiles) * 512 + lane * 16;
  const __bf16* bptr = Wp + ((long)(nb0 >> 4) + wn * 4) * 512 + lane * 16;
  const long bstep = (long)ntiles16 * 512;

  for (int kt = 0; kt < ktiles; ++kt) {
    const v16bf fa = *(const v16bf*)aptr;
#pragma unroll
    for (int i = 0; i < 4; ++i) {
      const v16bf fb = *(const v16bf*)(bptr + i * 512);
      acc[i] = __builtin_amdgcn_wmma_f32_16x16x32_bf16(
          false, fa, false, fb, (short)0, acc[i], false, false);
    }
    aptr += 512;
    bptr += bstep;
    __builtin_prefetch(bptr, 0, 2);     // hint next B k-block (L2 stream)
  }

  const int row0 = mb0 + wm * 16;
  if constexpr (OMODE == 0) {
    float* C = (float*)Cbase + bb * sCb + hh * sCh;
#pragma unroll
    for (int i = 0; i < 4; ++i) {
      const int col = nb0 + wn * 64 + i * 16 + lm;
      const float bv = (FLAGS & 1) ? biasBase[col] : 0.0f;
#pragma unroll
      for (int vv = 0; vv < 8; ++vv) {
        float val = acc[i][vv] + bv;
        if (FLAGS & 2) val = fmaxf(val, 0.0f);
        C[(long)(row0 + vv + hf * 8) * ldc + col] = val;
      }
    }
  } else if constexpr (OMODE == 1) {
    __bf16* C = (__bf16*)Cbase + bb * sCb + hh * sCh;
#pragma unroll
    for (int i = 0; i < 4; ++i) {
      const int col = nb0 + wn * 64 + i * 16 + lm;
#pragma unroll
      for (int vv = 0; vv < 8; ++vv)
        C[(long)(row0 + vv + hf * 8) * ldc + col] = f2bf(acc[i][vv]);
    }
  } else if constexpr (OMODE == 2) {
    // V-fragment-major: lane'=(dh&15)+16*((t>>4)&1), slot=t&15
    __bf16* C = (__bf16*)Cbase + bb * sCb + hh * sCh;
#pragma unroll
    for (int i = 0; i < 4; ++i) {
#pragma unroll
      for (int vv = 0; vv < 8; ++vv) {
        const int t = row0 + vv + hf * 8;
        const long off =
            (((long)(t >> 5) * 4 + i) * 32 + lm + (((t >> 4) & 1) << 4)) * 16 +
            (t & 15);
        C[off] = f2bf(acc[i][vv]);
      }
    }
  } else {
    // A-fragment-major packed output (feeds the next GEMM directly)
    __bf16* C = (__bf16*)Cbase + bb * sCb + hh * sCh;
#pragma unroll
    for (int i = 0; i < 4; ++i) {
      const int col = nb0 + wn * 64 + i * 16 + lm;
      const float bv = (FLAGS & 1) ? biasBase[col] : 0.0f;
#pragma unroll
      for (int vv = 0; vv < 8; ++vv) {
        float val = acc[i][vv] + bv;
        if (FLAGS & 2) val = fmaxf(val, 0.0f);
        C[apack_off(row0 + vv + hf * 8, col, N)] = f2bf(val);
      }
    }
  }
}

// -------------------------------------------------------------------------
// Fused CAUSAL attention for one (b,h) and one 16-query-row tile.
// Exploits causality: only key tiles nt <= qt are computed (~2x less work).
// q,k: bf16 row-major [S,DH]; vf: bf16 V-fragment-major. One wave per block.
// Output written directly in A-fragment-major bf16 (feeds the Wo GEMM).
// -------------------------------------------------------------------------
__global__ void attention_kernel(const __bf16* __restrict__ q,
                                 const __bf16* __restrict__ k,
                                 const __bf16* __restrict__ vf,
                                 __bf16* __restrict__ obf) {
  extern __shared__ float smem[];
  float* sc = smem;                                   // 16 * S f32 scores
  __bf16* pbf = (__bf16*)(smem + 16 * S_);            // 16 * S bf16 probs

  const int qt = blockIdx.x;
  const int z = blockIdx.y;
  const int b = z / H_, h = z % H_;
  const __bf16* Q = q + (long)z * S_ * DH_;
  const __bf16* Kp = k + (long)z * S_ * DH_;
  const __bf16* Vf = vf + (long)z * S_ * DH_;

  const int lane = threadIdx.x;
  const int hf = lane >> 4;
  const int lm = lane & 15;
  const int qr0 = qt * 16;
  const int valid = qr0 + 16;           // keys 0..valid-1 can be attended
  const int kslim = (valid + 31) >> 5;  // 32-wide k-blocks for P@V

  // ---- Q tile as two A fragments: direct 16-byte loads ----
  union FA { v16bf v; v8bf h[2]; };
  FA aq[2];
  const __bf16* qrow = Q + (long)(qr0 + lm) * DH_;
#pragma unroll
  for (int s2 = 0; s2 < 2; ++s2) {
    aq[s2].h[0] = *(const v8bf*)(qrow + s2 * 32 + hf * 16);
    aq[s2].h[1] = *(const v8bf*)(qrow + s2 * 32 + hf * 16 + 8);
  }
  const float scl = 0.125f;  // 1/sqrt(64)

  // ---- scores = Q @ K^T over the causal band only ----
  for (int nt = 0; nt <= qt; ++nt) {
    const __bf16* krow = Kp + (long)(nt * 16 + lm) * DH_;
    v8f c = (v8f){0.f,0.f,0.f,0.f,0.f,0.f,0.f,0.f};
#pragma unroll
    for (int s2 = 0; s2 < 2; ++s2) {
      const v16bf fb = *(const v16bf*)(krow + s2 * 32 + hf * 16);
      c = __builtin_amdgcn_wmma_f32_16x16x32_bf16(false, aq[s2].v, false, fb,
                                                  (short)0, c, false, false);
    }
#pragma unroll
    for (int vv = 0; vv < 8; ++vv) {
      int row = vv + hf * 8;
      int tcol = nt * 16 + lm;
      sc[row * S_ + tcol] = (tcol <= qr0 + row) ? c[vv] * scl : -3.0e38f;
    }
  }
  __syncthreads();

  // ---- softmax over [0, valid): 2 lanes/row, float4 passes ----
  {
    float* r = sc + lm * S_;
    __bf16* pr = pbf + lm * S_;
    const int vl = valid >> 1;          // per-lane span (multiple of 8)
    const int t0 = hf * vl, t1 = t0 + vl;
    float mx = -3.0e38f;
    for (int t = t0; t < t1; t += 4) {
      float4 v4 = *(const float4*)&r[t];
      mx = fmaxf(mx, fmaxf(fmaxf(v4.x, v4.y), fmaxf(v4.z, v4.w)));
    }
    mx = fmaxf(mx, __shfl_xor(mx, 16, 32));
    float sum = 0.0f;
    for (int t = t0; t < t1; t += 4) {
      float4 v4 = *(float4*)&r[t];
      v4.x = __expf(v4.x - mx); v4.y = __expf(v4.y - mx);
      v4.z = __expf(v4.z - mx); v4.w = __expf(v4.w - mx);
      sum += (v4.x + v4.y) + (v4.z + v4.w);
      *(float4*)&r[t] = v4;
    }
    sum += __shfl_xor(sum, 16, 32);
    const float inv = 1.0f / sum;
    for (int t = t0; t < t1; t += 4) {
      float4 v4 = *(const float4*)&r[t];
      union { __bf16 h[4]; uint2 u2; } p;
      p.h[0] = f2bf(v4.x * inv); p.h[1] = f2bf(v4.y * inv);
      p.h[2] = f2bf(v4.z * inv); p.h[3] = f2bf(v4.w * inv);
      *(uint2*)&pr[t] = p.u2;
    }
    // zero-fill the tail half-block so the last P@V WMMA contributes 0
    if (valid < kslim * 32) {
      uint4 zz = {0u, 0u, 0u, 0u};
      *(uint4*)&pr[valid + hf * 8] = zz;
    }
  }
  __syncthreads();

  // ---- out = P @ V over the causal band: barrier-free inner loop ----
  v8f acc[4];
#pragma unroll
  for (int i = 0; i < 4; ++i) acc[i] = (v8f){0.f,0.f,0.f,0.f,0.f,0.f,0.f,0.f};

  const __bf16* prow = pbf + lm * S_;
  for (int ks = 0; ks < kslim; ++ks) {
    const int kk0 = ks * 32;
    FA fa;
    fa.h[0] = *(const v8bf*)(prow + kk0 + hf * 16);
    fa.h[1] = *(const v8bf*)(prow + kk0 + hf * 16 + 8);
#pragma unroll
    for (int i = 0; i < 4; ++i) {
      const v16bf fb = *(const v16bf*)&Vf[(((long)ks * 4 + i) * 32 + lane) * 16];
      acc[i] = __builtin_amdgcn_wmma_f32_16x16x32_bf16(false, fa.v, false, fb,
                                                       (short)0, acc[i], false,
                                                       false);
    }
  }
  // ---- write output directly A-fragment-major (cols of [B*S, DM]) ----
#pragma unroll
  for (int i = 0; i < 4; ++i) {
    const int c = h * DH_ + i * 16 + lm;
#pragma unroll
    for (int vv = 0; vv < 8; ++vv) {
      const int m = b * S_ + qr0 + vv + hf * 8;
      obf[apack_off(m, c, DM_)] = f2bf(acc[i][vv]);
    }
  }
}

// -------------------------------------------------------------------------
// x = LayerNorm(x + add): unbiased std (ddof=1), divide by (std+eps).
// One block (256 thr) per token row; each thread owns 4 consecutive dims.
// Writes f32 x (residual chain) + packed bf16 xbf (next GEMM's A operand).
// -------------------------------------------------------------------------
__global__ void ln_residual_kernel(float* __restrict__ x,
                                   __bf16* __restrict__ xbf,
                                   const float* __restrict__ add,
                                   const float* __restrict__ g,
                                   const float* __restrict__ beta) {
  const int row = blockIdx.x;
  __shared__ float red[256];
  const int d0 = threadIdx.x * 4;
  float4 xv = *(const float4*)&x[(long)row * DM_ + d0];
  float4 av = *(const float4*)&add[(long)row * DM_ + d0];
  float vals[4] = {xv.x + av.x, xv.y + av.y, xv.z + av.z, xv.w + av.w};

  red[threadIdx.x] = (vals[0] + vals[1]) + (vals[2] + vals[3]);
  __syncthreads();
  for (int off = 128; off > 0; off >>= 1) {
    if (threadIdx.x < off) red[threadIdx.x] += red[threadIdx.x + off];
    __syncthreads();
  }
  const float mean = red[0] / (float)DM_;
  __syncthreads();
  float lss = 0.0f;
#pragma unroll
  for (int i = 0; i < 4; ++i) {
    float d = vals[i] - mean;
    lss += d * d;
  }
  red[threadIdx.x] = lss;
  __syncthreads();
  for (int off = 128; off > 0; off >>= 1) {
    if (threadIdx.x < off) red[threadIdx.x] += red[threadIdx.x + off];
    __syncthreads();
  }
  const float sd = sqrtf(red[0] / (float)(DM_ - 1));
  const float inv = 1.0f / (sd + 1e-6f);
  const float4 gv = *(const float4*)&g[d0];
  const float4 bv = *(const float4*)&beta[d0];
  float4 r;
  r.x = gv.x * (vals[0] - mean) * inv + bv.x;
  r.y = gv.y * (vals[1] - mean) * inv + bv.y;
  r.z = gv.z * (vals[2] - mean) * inv + bv.z;
  r.w = gv.w * (vals[3] - mean) * inv + bv.w;
  *(float4*)&x[(long)row * DM_ + d0] = r;
  union { __bf16 h[4]; uint2 u2; } p;
  p.h[0] = f2bf(r.x); p.h[1] = f2bf(r.y); p.h[2] = f2bf(r.z); p.h[3] = f2bf(r.w);
  *(uint2*)&xbf[apack_off(row, d0, DM_)] = p.u2;
}

// -------------------------------------------------------------------------
extern "C" void kernel_launch(void* const* d_in, const int* in_sizes, int n_in,
                              void* d_out, int out_size, void* d_ws, size_t ws_size,
                              hipStream_t stream) {
  (void)in_sizes; (void)n_in; (void)out_size; (void)ws_size;
  const int*   tokens = (const int*)d_in[0];
  const float* emb    = (const float*)d_in[1];
  const float* Wq     = (const float*)d_in[2];
  const float* Wk     = (const float*)d_in[3];
  const float* Wv     = (const float*)d_in[4];
  const float* Wo     = (const float*)d_in[5];
  const float* bo     = (const float*)d_in[6];
  const float* ln1_g  = (const float*)d_in[7];
  const float* ln1_b  = (const float*)d_in[8];
  const float* W1     = (const float*)d_in[9];
  const float* b1     = (const float*)d_in[10];
  const float* W2     = (const float*)d_in[11];
  const float* b2     = (const float*)d_in[12];
  const float* ln2_g  = (const float*)d_in[13];
  const float* ln2_b  = (const float*)d_in[14];
  const float* Wout   = (const float*)d_in[15];
  const float* bout   = (const float*)d_in[16];
  float* out = (float*)d_out;

  // Workspace carve-out
  float* ws = (float*)d_ws;
  const size_t xsz = (size_t)B_ * S_ * DM_;        // 2,097,152 elems
  const size_t hsz = (size_t)B_ * S_ * DFF_;       // 8,388,608 elems
  float* x    = ws;                        // f32 [B,S,DM] residual stream
  float* tmp  = x + xsz;                   // f32 [B,S,DM] proj / ffn2 output
  __bf16* xbf = (__bf16*)(tmp + xsz);      // packed A-major copy of x
  __bf16* obf = xbf + xsz;                 // packed attention output
  __bf16* hbf = obf + xsz;                 // packed ffn1 output
  __bf16* qp  = hbf + hsz;                 // bf16 row-major Q
  __bf16* kp  = qp + xsz;                  // bf16 row-major K
  __bf16* vfp = kp + xsz;                  // bf16 V-fragment-major
  __bf16* wp  = vfp + xsz;                 // repacked weights (<= 65.5 MB)

  // 1) Embedding + positional encoding
  embed_kernel<<<dim3(B_ * S_), 256, 0, stream>>>(tokens, emb, x, xbf);

  const long sDMDH = (long)DM_ * DH_;
  const size_t attn_lds = 16 * S_ * sizeof(float) + 16 * S_ * sizeof(__bf16);
  auto rpgrid = [](int K, int N, int nb) {
    return dim3((N + 255) / 256, K / 4, nb);
  };

  for (int l = 0; l < L_; ++l) {
    const float* Wql = Wq + (size_t)l * H_ * DM_ * DH_;
    const float* Wkl = Wk + (size_t)l * H_ * DM_ * DH_;
    const float* Wvl = Wv + (size_t)l * H_ * DM_ * DH_;
    const float* Wol = Wo + (size_t)l * DM_ * DM_;
    const float* bol = bo + (size_t)l * DM_;
    const float* W1l = W1 + (size_t)l * DM_ * DFF_;
    const float* b1l = b1 + (size_t)l * DFF_;
    const float* W2l = W2 + (size_t)l * DFF_ * DM_;
    const float* b2l = b2 + (size_t)l * DM_;

    // ---- Q/K/V projections (WMW=4: 64x64 tiles, N=64 exact) ----
    dim3 gq(1, S_ / 64, B_ * H_);
    dim3 rq = rpgrid(DM_, DH_, H_);
    repack_w_kernel<<<rq, 256, 0, stream>>>(Wql, wp, DH_, sDMDH, sDMDH);
    gemm_bf16_wmma<4, 0, 1><<<gq, 128, 0, stream>>>(xbf, wp, nullptr, qp,
        DH_, DM_, DH_,
        (long)S_ * DM_, 0L, sDMDH, (long)H_ * S_ * DH_, (long)S_ * DH_, H_);
    repack_w_kernel<<<rq, 256, 0, stream>>>(Wkl, wp, DH_, sDMDH, sDMDH);
    gemm_bf16_wmma<4, 0, 1><<<gq, 128, 0, stream>>>(xbf, wp, nullptr, kp,
        DH_, DM_, DH_,
        (long)S_ * DM_, 0L, sDMDH, (long)H_ * S_ * DH_, (long)S_ * DH_, H_);
    repack_w_kernel<<<rq, 256, 0, stream>>>(Wvl, wp, DH_, sDMDH, sDMDH);
    gemm_bf16_wmma<4, 0, 2><<<gq, 128, 0, stream>>>(xbf, wp, nullptr, vfp,
        DH_, DM_, DH_,
        (long)S_ * DM_, 0L, sDMDH, (long)H_ * S_ * DH_, (long)S_ * DH_, H_);

    // ---- Fused causal attention (96 KB dynamic LDS) ----
    attention_kernel<<<dim3(S_ / 16, B_ * H_), 32, attn_lds, stream>>>(
        qp, kp, vfp, obf);

    // ---- Output projection: tmp = obf @ Wo + bo ----
    repack_w_kernel<<<rpgrid(DM_, DM_, 1), 256, 0, stream>>>(Wol, wp, DM_, 0L, 0L);
    gemm_bf16_wmma<2, 1, 0><<<dim3(DM_ / 128, (B_ * S_) / 32, 1), 128, 0, stream>>>(
        obf, wp, bol, tmp, DM_, DM_, DM_,
        0L, 0L, 0L, 0L, 0L, 1);

    // ---- x = LN(tmp + x) ----
    ln_residual_kernel<<<dim3(B_ * S_), 256, 0, stream>>>(
        x, xbf, tmp, ln1_g + (size_t)l * DM_, ln1_b + (size_t)l * DM_);

    // ---- FFN: hbf = relu(x @ W1 + b1) packed; tmp = hbf @ W2 + b2 ----
    repack_w_kernel<<<rpgrid(DM_, DFF_, 1), 256, 0, stream>>>(W1l, wp, DFF_, 0L, 0L);
    gemm_bf16_wmma<2, 3, 3><<<dim3(DFF_ / 128, (B_ * S_) / 32, 1), 128, 0, stream>>>(
        xbf, wp, b1l, hbf, DFF_, DM_, DFF_,
        0L, 0L, 0L, 0L, 0L, 1);
    repack_w_kernel<<<rpgrid(DFF_, DM_, 1), 256, 0, stream>>>(W2l, wp, DM_, 0L, 0L);
    gemm_bf16_wmma<2, 1, 0><<<dim3(DM_ / 128, (B_ * S_) / 32, 1), 128, 0, stream>>>(
        hbf, wp, b2l, tmp, DM_, DFF_, DM_,
        0L, 0L, 0L, 0L, 0L, 1);

    // ---- x = LN(tmp + x) ----
    ln_residual_kernel<<<dim3(B_ * S_), 256, 0, stream>>>(
        x, xbf, tmp, ln2_g + (size_t)l * DM_, ln2_b + (size_t)l * DM_);
  }

  // ---- Final logits: out = x @ Wout + bout  (2048 x 32000 x 1024) ----
  repack_w_kernel<<<rpgrid(DM_, V_, 1), 256, 0, stream>>>(Wout, wp, V_, 0L, 0L);
  gemm_bf16_wmma<2, 1, 0><<<dim3(V_ / 128, (B_ * S_) / 32, 1), 128, 0, stream>>>(
      xbf, wp, bout, out, V_, DM_, V_,
      0L, 0L, 0L, 0L, 0L, 1);
}
